// Seq2SeqModel_64158221468102
// MI455X (gfx1250) — matrix operand order repaired and checked by reading, not compiled, not run on hardware
//
#include <hip/hip_runtime.h>

#define VOC   32000
#define EMB   256
#define HID   512
#define BATCH 32
#define SSRC  128
#define STGT  64

typedef __bf16 bf16_t;
typedef bf16_t v16bf __attribute__((ext_vector_type(16)));
typedef float  v8f   __attribute__((ext_vector_type(8)));

union BFrag { v16bf v; uint4 u4[2]; };
union CFrag { v8f   v; float f[8]; };

static __device__ __forceinline__ unsigned short f2bf(float x) {
    unsigned int u = __float_as_uint(x);
    u += 0x7FFFu + ((u >> 16) & 1u);          // round-to-nearest-even
    return (unsigned short)(u >> 16);
}
static __device__ __forceinline__ float bf2f(unsigned short s) {
    return __uint_as_float(((unsigned int)s) << 16);
}
static __device__ __forceinline__ float sigf(float x) { return 1.0f / (1.0f + __expf(-x)); }

// ---------------------------------------------------------------------------
// Weight swizzle: Bmat[k][n] = k<k1 ? W1[n*ld1+k] : W2[n*ld2+(k-k1)], emitted
// in this file's contiguous-k bf16 B-fragment order (A loads match): tile(kt,nt) -> 32 lanes x 16 halves contiguous,
// lane l holds N=nt*16+(l&15), K = kt*32 + (l>>4)*16 + h
// ---------------------------------------------------------------------------
__global__ void swizzle_w(const float* __restrict__ W1, int ld1, int k1,
                          const float* __restrict__ W2, int ld2,
                          unsigned short* __restrict__ out, int K, int N)
{
    size_t e = (size_t)blockIdx.x * blockDim.x + threadIdx.x;
    size_t total = (size_t)K * (size_t)N;
    if (e >= total) return;
    int h    = (int)(e & 15);
    int lane = (int)((e >> 4) & 31);
    size_t tile = e >> 9;
    int ntile16 = N >> 4;
    int nt = (int)(tile % ntile16);
    int kt = (int)(tile / ntile16);
    int k = kt * 32 + (lane >> 4) * 16 + h;
    int n = nt * 16 + (lane & 15);
    float val = (k < k1) ? W1[(size_t)n * ld1 + k] : W2[(size_t)n * ld2 + (k - k1)];
    out[e] = f2bf(val);
}

// Embedding gather -> bf16, layout (S, B, E)
__global__ void gather_emb(const int* __restrict__ idx, int S,
                           const float* __restrict__ emb,
                           unsigned short* __restrict__ out, size_t total)
{
    size_t e = (size_t)blockIdx.x * blockDim.x + threadIdx.x;
    if (e >= total) return;
    int d = (int)(e & (EMB - 1));
    int b = (int)((e >> 8) & (BATCH - 1));
    int s = (int)(e >> 13);
    int tok = idx[b * S + s];
    out[e] = f2bf(emb[(size_t)tok * EMB + d]);
}

// ---------------------------------------------------------------------------
// bf16 WMMA GEMM: C(MxN,f32) = A(MxK,bf16,row-major lda) @ Bsw(KxN swizzled)
//                 + bias0[n] + bias1[n].  One wave = 16x32 tile (2 WMMAs/K-step).
// ---------------------------------------------------------------------------
__global__ void gemm_bf16_wmma(const unsigned short* __restrict__ A, int lda,
                               const unsigned short* __restrict__ Bsw,
                               const float* __restrict__ bias0,
                               const float* __restrict__ bias1,
                               float* __restrict__ C, long long ldc,
                               int M, int N, int K)
{
    const int lane = threadIdx.x & 31;
    const int wid  = blockIdx.x * (blockDim.x >> 5) + (threadIdx.x >> 5);
    const int ntiles2 = N >> 5;
    const int nwaves  = (M >> 4) * ntiles2;
    if (wid >= nwaves) return;
    const int mt  = wid / ntiles2;
    const int nt2 = wid - mt * ntiles2;
    const int hs   = lane >> 4;      // half-select
    const int lrow = lane & 15;

    BFrag a, b0, b1;
    CFrag c0, c1;
    #pragma unroll
    for (int r = 0; r < 8; ++r) { c0.f[r] = 0.0f; c1.f[r] = 0.0f; }

    const unsigned short* arow  = A + (size_t)(mt * 16 + lrow) * lda + hs * 16;
    const int ntile16 = N >> 4;
    const unsigned short* bbase = Bsw + ((size_t)(nt2 * 2) * 32 + lane) * 16;

    for (int kt = 0; kt < (K >> 5); ++kt) {
        const uint4* pa = reinterpret_cast<const uint4*>(arow + kt * 32);
        a.u4[0] = pa[0];               // K = kb+0..7
        a.u4[1] = pa[1];               // K = kb+8..15
        const unsigned short* bp = bbase + (size_t)kt * ntile16 * 512;
        const uint4* pb0 = reinterpret_cast<const uint4*>(bp);
        b0.u4[0] = pb0[0]; b0.u4[1] = pb0[1];
        const uint4* pb1 = reinterpret_cast<const uint4*>(bp + 512);
        b1.u4[0] = pb1[0]; b1.u4[1] = pb1[1];
        c0.v = __builtin_amdgcn_wmma_f32_16x16x32_bf16(false, a.v, false, b0.v,
                                                       (short)0, c0.v, false, false);
        c1.v = __builtin_amdgcn_wmma_f32_16x16x32_bf16(false, a.v, false, b1.v,
                                                       (short)0, c1.v, false, false);
    }

    const int col0 = nt2 * 32 + lrow;
    const float bb0 = (bias0 ? bias0[col0]      : 0.0f) + (bias1 ? bias1[col0]      : 0.0f);
    const float bb1 = (bias0 ? bias0[col0 + 16] : 0.0f) + (bias1 ? bias1[col0 + 16] : 0.0f);
    #pragma unroll
    for (int r = 0; r < 8; ++r) {
        const int row = mt * 16 + hs * 8 + r;
        C[(size_t)row * ldc + col0]      = c0.f[r] + bb0;
        C[(size_t)row * ldc + col0 + 16] = c1.f[r] + bb1;
    }
}

// LSTM gate nonlinearity: g(32x2048) -> c,h ; also emit bf16 h to 1-2 dests
__global__ void lstm_gates(const float* __restrict__ g,
                           float* __restrict__ c, float* __restrict__ h,
                           unsigned short* __restrict__ dst0, int s0,
                           unsigned short* __restrict__ dst1, int s1)
{
    int i = blockIdx.x * blockDim.x + threadIdx.x;
    if (i >= BATCH * HID) return;
    int b = i >> 9, j = i & (HID - 1);
    const float* gr = g + (size_t)b * 4 * HID;
    float iv = gr[j], fv = gr[HID + j], gv = gr[2 * HID + j], ov = gr[3 * HID + j];
    float cv = sigf(fv) * c[i] + sigf(iv) * tanhf(gv);
    float hv = sigf(ov) * tanhf(cv);
    c[i] = cv; h[i] = hv;
    unsigned short hb = f2bf(hv);
    dst0[(size_t)b * s0 + j] = hb;
    if (dst1) dst1[(size_t)b * s1 + j] = hb;
}

// Attention (per-batch workgroup): q = h@Wq^T ; scores = v.tanh(q+enc_proj) ;
// softmax ; context = attw @ enc_out -> bf16 into act_dec slot.
__global__ void attention(const float* __restrict__ h,
                          const float* __restrict__ attW,   // (512,1536) rows: [Wq|We]
                          const float* __restrict__ vvec,
                          const float* __restrict__ enc_proj, // (B,S,512) f32
                          const unsigned short* __restrict__ enc_out, // (B,S,1024) bf16
                          unsigned short* __restrict__ ctx_dst, int ctx_stride)
{
    __shared__ float q[HID];
    __shared__ float sc[SSRC];
    __shared__ float wt[SSRC];
    __shared__ float red[2];
    const int b = blockIdx.x, tid = threadIdx.x;

    for (int j = tid; j < HID; j += blockDim.x) {
        float s = 0.0f;
        const float* hr = h + b * HID;
        const float* wr = attW + (size_t)j * (3 * HID);
        for (int k = 0; k < HID; ++k) s += hr[k] * wr[k];
        q[j] = s;
    }
    __syncthreads();
    if (tid < SSRC) {
        const float* ep = enc_proj + ((size_t)b * SSRC + tid) * HID;
        float s = 0.0f;
        for (int j = 0; j < HID; ++j) s += vvec[j] * tanhf(q[j] + ep[j]);
        sc[tid] = s;
    }
    __syncthreads();
    if (tid == 0) {
        float m = sc[0];
        for (int s = 1; s < SSRC; ++s) m = fmaxf(m, sc[s]);
        red[0] = m;
    }
    __syncthreads();
    if (tid < SSRC) wt[tid] = __expf(sc[tid] - red[0]);
    __syncthreads();
    if (tid == 0) {
        float s = 0.0f;
        for (int k = 0; k < SSRC; ++k) s += wt[k];
        red[1] = 1.0f / s;
    }
    __syncthreads();
    const float inv = red[1];
    for (int d = tid; d < 2 * HID; d += blockDim.x) {
        float acc = 0.0f;
        const unsigned short* eo = enc_out + (size_t)b * SSRC * (2 * HID) + d;
        for (int s = 0; s < SSRC; ++s) acc += wt[s] * inv * bf2f(eo[(size_t)s * (2 * HID)]);
        ctx_dst[(size_t)b * ctx_stride + d] = f2bf(acc);
    }
}

__global__ void pack_x(const unsigned short* __restrict__ xsrc,  // step slice (32,256)
                       unsigned short* __restrict__ act, int stride)
{
    int i = blockIdx.x * blockDim.x + threadIdx.x;
    if (i >= BATCH * EMB) return;
    act[(size_t)(i >> 8) * stride + (i & (EMB - 1))] = xsrc[i];
}

__global__ void h_to_bf(const float* __restrict__ h, unsigned short* __restrict__ dst, int stride)
{
    int i = blockIdx.x * blockDim.x + threadIdx.x;
    if (i >= BATCH * HID) return;
    dst[(size_t)(i >> 9) * stride + (i & (HID - 1))] = f2bf(h[i]);
}

__global__ void zero_f32(float* __restrict__ p, int n)
{
    int i = blockIdx.x * blockDim.x + threadIdx.x;
    if (i < n) p[i] = 0.0f;
}

__global__ void zero_bf_rows(unsigned short* __restrict__ p, int rowlen, int stride, int rows)
{
    int i = blockIdx.x * blockDim.x + threadIdx.x;
    if (i >= rowlen * rows) return;
    p[(size_t)(i / rowlen) * stride + (i % rowlen)] = 0;
}

// ---------------------------------------------------------------------------
extern "C" void kernel_launch(void* const* d_in, const int* in_sizes, int n_in,
                              void* d_out, int out_size, void* d_ws, size_t ws_size,
                              hipStream_t stream)
{
    (void)in_sizes; (void)n_in; (void)out_size; (void)ws_size;
    const int*   src       = (const int*)  d_in[0];
    const int*   tgt       = (const int*)  d_in[1];
    const float* emb       = (const float*)d_in[2];
    const float* eWih_f    = (const float*)d_in[3];
    const float* eWhh_f    = (const float*)d_in[4];
    const float* ebih_f    = (const float*)d_in[5];
    const float* ebhh_f    = (const float*)d_in[6];
    const float* eWih_b    = (const float*)d_in[7];
    const float* eWhh_b    = (const float*)d_in[8];
    const float* ebih_b    = (const float*)d_in[9];
    const float* ebhh_b    = (const float*)d_in[10];
    const float* dWih      = (const float*)d_in[11];
    const float* dWhh      = (const float*)d_in[12];
    const float* dbih      = (const float*)d_in[13];
    const float* dbhh      = (const float*)d_in[14];
    const float* attW      = (const float*)d_in[15];
    const float* attb      = (const float*)d_in[16];
    const float* vvec      = (const float*)d_in[17];
    const float* outW      = (const float*)d_in[18];
    const float* outb      = (const float*)d_in[19];
    float* out = (float*)d_out;

    char* w = (char*)d_ws; size_t off = 0;
    auto alloc = [&](size_t bytes) -> void* {
        void* p = w + off; off += (bytes + 255) & ~(size_t)255; return p;
    };
    unsigned short* src_emb  = (unsigned short*)alloc((size_t)SSRC * BATCH * EMB * 2);
    unsigned short* tgt_emb  = (unsigned short*)alloc((size_t)STGT * BATCH * EMB * 2);
    unsigned short* encF_sw  = (unsigned short*)alloc((size_t)768  * 2048 * 2);
    unsigned short* encB_sw  = (unsigned short*)alloc((size_t)768  * 2048 * 2);
    unsigned short* dec_sw   = (unsigned short*)alloc((size_t)1792 * 2048 * 2);
    unsigned short* We_sw    = (unsigned short*)alloc((size_t)1024 * 512  * 2);
    unsigned short* outW_sw  = (unsigned short*)alloc((size_t)512  * VOC  * 2);
    unsigned short* enc_out  = (unsigned short*)alloc((size_t)BATCH * SSRC * 1024 * 2);
    float*          enc_proj = (float*)alloc((size_t)BATCH * SSRC * HID * 4);
    float*          gbuf     = (float*)alloc((size_t)BATCH * 2048 * 4);
    float*          hbuf     = (float*)alloc((size_t)BATCH * HID * 4);
    float*          cbuf     = (float*)alloc((size_t)BATCH * HID * 4);
    unsigned short* act_enc  = (unsigned short*)alloc((size_t)BATCH * 768  * 2);
    unsigned short* act_dec  = (unsigned short*)alloc((size_t)BATCH * 1792 * 2);

    const int T = 256;
    auto blk = [&](size_t n) { return (unsigned)((n + T - 1) / T); };
    auto gemm = [&](const unsigned short* A, int lda, const unsigned short* B,
                    const float* b0, const float* b1, float* C, long long ldc,
                    int M, int N, int K) {
        int waves = (M >> 4) * (N >> 5);
        gemm_bf16_wmma<<<(waves + 3) / 4, 128, 0, stream>>>(A, lda, B, b0, b1, C, ldc, M, N, K);
    };

    // ---- prep: embeddings + swizzled bf16 weights ---------------------------
    gather_emb<<<blk((size_t)SSRC*BATCH*EMB), T, 0, stream>>>(src, SSRC, emb, src_emb, (size_t)SSRC*BATCH*EMB);
    gather_emb<<<blk((size_t)STGT*BATCH*EMB), T, 0, stream>>>(tgt, STGT, emb, tgt_emb, (size_t)STGT*BATCH*EMB);
    swizzle_w<<<blk((size_t)768*2048),  T, 0, stream>>>(eWih_f, EMB, EMB, eWhh_f, HID, encF_sw, 768, 2048);
    swizzle_w<<<blk((size_t)768*2048),  T, 0, stream>>>(eWih_b, EMB, EMB, eWhh_b, HID, encB_sw, 768, 2048);
    swizzle_w<<<blk((size_t)1792*2048), T, 0, stream>>>(dWih, 1280, 1280, dWhh, HID, dec_sw, 1792, 2048);
    swizzle_w<<<blk((size_t)1024*512),  T, 0, stream>>>(attW + HID, 3*HID, 1024, nullptr, 0, We_sw, 1024, 512);
    swizzle_w<<<blk((size_t)512*VOC),   T, 0, stream>>>(outW, HID, HID, nullptr, 0, outW_sw, 512, VOC);

    // ---- encoder forward ----------------------------------------------------
    zero_f32<<<blk(BATCH*HID), T, 0, stream>>>(hbuf, BATCH*HID);
    zero_f32<<<blk(BATCH*HID), T, 0, stream>>>(cbuf, BATCH*HID);
    zero_bf_rows<<<blk(BATCH*HID), T, 0, stream>>>(act_enc + EMB, HID, 768, BATCH);
    for (int t = 0; t < SSRC; ++t) {
        pack_x<<<blk(BATCH*EMB), T, 0, stream>>>(src_emb + (size_t)t*BATCH*EMB, act_enc, 768);
        gemm(act_enc, 768, encF_sw, ebih_f, ebhh_f, gbuf, 2048, BATCH, 2048, 768);
        lstm_gates<<<blk(BATCH*HID), T, 0, stream>>>(gbuf, cbuf, hbuf,
            act_enc + EMB, 768, enc_out + (size_t)t*1024, SSRC*1024);
    }
    // ---- encoder backward ---------------------------------------------------
    zero_f32<<<blk(BATCH*HID), T, 0, stream>>>(hbuf, BATCH*HID);
    zero_f32<<<blk(BATCH*HID), T, 0, stream>>>(cbuf, BATCH*HID);
    zero_bf_rows<<<blk(BATCH*HID), T, 0, stream>>>(act_enc + EMB, HID, 768, BATCH);
    for (int tb = 0; tb < SSRC; ++tb) {
        int s = SSRC - 1 - tb;
        pack_x<<<blk(BATCH*EMB), T, 0, stream>>>(src_emb + (size_t)s*BATCH*EMB, act_enc, 768);
        gemm(act_enc, 768, encB_sw, ebih_b, ebhh_b, gbuf, 2048, BATCH, 2048, 768);
        lstm_gates<<<blk(BATCH*HID), T, 0, stream>>>(gbuf, cbuf, hbuf,
            act_enc + EMB, 768, enc_out + (size_t)s*1024 + HID, SSRC*1024);
    }
    // ---- enc_proj = enc_out @ We^T + att_b ----------------------------------
    gemm(enc_out, 1024, We_sw, attb, nullptr, enc_proj, HID, BATCH*SSRC, HID, 1024);

    // ---- decoder (h,c continue from backward-encoder final state) -----------
    h_to_bf<<<blk(BATCH*HID), T, 0, stream>>>(hbuf, act_dec + EMB + 2*HID, 1792);
    for (int t = 0; t < STGT; ++t) {
        pack_x<<<blk(BATCH*EMB), T, 0, stream>>>(tgt_emb + (size_t)t*BATCH*EMB, act_dec, 1792);
        attention<<<BATCH, 256, 0, stream>>>(hbuf, attW, vvec, enc_proj, enc_out,
                                             act_dec + EMB, 1792);
        gemm(act_dec, 1792, dec_sw, dbih, dbhh, gbuf, 2048, BATCH, 2048, 1792);
        lstm_gates<<<blk(BATCH*HID), T, 0, stream>>>(gbuf, cbuf, hbuf,
            act_dec + EMB + 2*HID, 1792, nullptr, 0);
        // logits: h(32x512) @ outW^T (512x32000) + out_b  -> out[:, t, :]
        gemm(act_dec + EMB + 2*HID, 1792, outW_sw, outb, nullptr,
             out + (size_t)t * VOC, (long long)STGT * VOC, BATCH, VOC, HID);
    }
}
